// RelationalEncoder_83880711291080
// MI455X (gfx1250) — compile-verified, matrix operand order checked by reading, not run on hardware
//
#include <hip/hip_runtime.h>

typedef _Float16 v8h  __attribute__((ext_vector_type(8)));
typedef _Float16 v16h __attribute__((ext_vector_type(16)));
typedef float    v8f  __attribute__((ext_vector_type(8)));

#define ROWS    256
#define THREADS 256
#define SROW    136   // padded LDS row stride in f16: 272B -> bank step 4, conflict-free b128 frags

// Load a 16-f16 WMMA fragment for one matrix row, chunk base kb.
// Per CDNA5 ISA 16-bit A 16x32 layout: lane<16 -> K 0..7 & 16..23,
// lane>=16 -> K 8..15 & 24..31 (kb pre-adds the 8*hi16 sub-offset).
__device__ __forceinline__ v16h frag16(const _Float16* rowp, int kb) {
  v8h lo = *(const v8h*)(rowp + kb);
  v8h hi = *(const v8h*)(rowp + kb + 16);
  return __builtin_shufflevector(lo, hi, 0,1,2,3,4,5,6,7,8,9,10,11,12,13,14,15);
}

#define WMMA_F16(A, B, C) \
  __builtin_amdgcn_wmma_f32_16x16x32_f16(false, (A), false, (B), (short)0, (C), false, false)

__global__ __launch_bounds__(THREADS)
void relenc_fused(const float* __restrict__ pos, const float* __restrict__ vel,
                  const float* __restrict__ att,
                  const float* __restrict__ W1, const float* __restrict__ b1v,
                  const float* __restrict__ rms_w,
                  const float* __restrict__ W2, const float* __restrict__ b2v,
                  const float* __restrict__ Wa, const float* __restrict__ bav,
                  float* __restrict__ out)
{
  __shared__ __align__(16) _Float16 sW[256 * SROW]; // transposed weights (reused per stage)
  __shared__ __align__(16) _Float16 sA[256 * SROW]; // features (k<64), then h2 (k<128)
  __shared__ __align__(16) _Float16 sB[256 * SROW]; // h1
  __shared__ float sBias[256];
  __shared__ float sRms[128];

  const int tid  = threadIdx.x;
  const int lane = tid & 31;
  const int wave = tid >> 5;
  const int hi16 = (lane >> 4);     // 0: lanes 0-15, 1: lanes 16-31
  const int l16  = lane & 15;
  const int koff = hi16 * 8;        // fragment K sub-offset

  // ---------------- Stage 0: W1^T into LDS (coalesced reads), biases, features ----------------
  for (int idx = tid; idx < 64 * 128; idx += THREADS) {
    int k = idx >> 7, n = idx & 127;          // consecutive lanes -> consecutive n: coalesced
    sW[n * SROW + k] = (_Float16)W1[idx];     // W1[idx] == W1[k*128 + n]
  }
  for (int n = tid; n < 128; n += THREADS) { sBias[n] = b1v[n]; sRms[n] = rms_w[n]; }

  // Warm L2 for later-stage weights while stage 1 runs.
  if (lane == 0) {
    __builtin_prefetch(W2 + wave * 2048, 0, 1);
    __builtin_prefetch(Wa + wave * 4096, 0, 1);
  }

  {
    const int r = tid;
    const long long pairIdx = (long long)blockIdx.x * ROWS + r;
    const int bt = (int)(pairIdx >> 10);
    const int ij = (int)(pairIdx & 1023);
    const int i = ij >> 5, j = ij & 31;
    const int bi = (bt * 32 + i) * 2, bj = (bt * 32 + j) * 2;

    float dx = pos[bi]     - pos[bj];
    float dy = pos[bi + 1] - pos[bj + 1];
    dx -= rintf(dx * (1.0f / 1024.0f)) * 1024.0f;
    dy -= rintf(dy * (1.0f / 1024.0f)) * 1024.0f;

    float dvx = vel[bi] - vel[bj], dvy = vel[bi + 1] - vel[bj + 1];
    float dist = sqrtf(dx * dx + dy * dy) + 1e-6f;
    float invd = 1.0f / dist;
    float dirx = dx * invd, diry = dy * invd;
    float rel  = sqrtf(dvx * dvx + dvy * dvy);
    float clos = (dvx * dx + dvy * dy) * invd;
    float inv_dist = 1.0f / (dist + 0.1f);
    float logd = __logf(dist + 1.0f);
    float tti  = fminf(dist / fmaxf(clos, 0.001f), 1.0e4f);
    float aix = att[bi], aiy = att[bi + 1], ajx = att[bj], ajy = att[bj + 1];
    float cos_ata =  dirx * aix + diry * aiy;
    float sin_ata =  dirx * aiy - diry * aix;
    float cos_aa  = -(dirx * ajx + diry * ajy);
    float sin_aa  = -dirx * ajy + diry * ajx;
    float cos_hca =  aix * ajx + aiy * ajy;
    float sin_hca =  aix * ajy - aiy * ajx;

    _Float16* rowp = sA + r * SROW;
    rowp[0]=(_Float16)dx;       rowp[1]=(_Float16)dy;
    rowp[2]=(_Float16)dvx;      rowp[3]=(_Float16)dvy;
    rowp[4]=(_Float16)dist;     rowp[5]=(_Float16)inv_dist;
    rowp[6]=(_Float16)rel;      rowp[7]=(_Float16)clos;
    rowp[8]=(_Float16)dirx;     rowp[9]=(_Float16)diry;
    rowp[10]=(_Float16)logd;    rowp[11]=(_Float16)tti;
    rowp[12]=(_Float16)cos_ata; rowp[13]=(_Float16)sin_ata;
    rowp[14]=(_Float16)cos_aa;  rowp[15]=(_Float16)sin_aa;
    rowp[16]=(_Float16)cos_hca; rowp[17]=(_Float16)sin_hca;

    float fx = dx * (6.283185307179586f / 1024.0f);
    float fy = dy * (6.283185307179586f / 1024.0f);
    #pragma unroll
    for (int bnd = 0; bnd < 8; ++bnd) {
      rowp[18 + bnd * 4 + 0] = (_Float16)__sinf(fx);
      rowp[18 + bnd * 4 + 1] = (_Float16)__sinf(fy);
      rowp[18 + bnd * 4 + 2] = (_Float16)__cosf(fx);
      rowp[18 + bnd * 4 + 3] = (_Float16)__cosf(fy);
      fx *= 2.0f; fy *= 2.0f;
    }
    #pragma unroll
    for (int c = 50; c < 64; ++c) rowp[c] = (_Float16)0.0f;
  }
  __syncthreads();

  // ---------------- GEMM1: h1 = feats(256x64) @ W1 + b1 -> sB (f16) ----------------
  {
    v16h a[2][2];
    #pragma unroll
    for (int mt = 0; mt < 2; ++mt) {
      const _Float16* arow = sA + (wave * 32 + mt * 16 + l16) * SROW;
      a[mt][0] = frag16(arow, koff);
      a[mt][1] = frag16(arow, 32 + koff);
    }
    #pragma unroll
    for (int nt = 0; nt < 8; ++nt) {
      const _Float16* brow = sW + (nt * 16 + l16) * SROW;
      v16h w0  = frag16(brow, koff);
      v16h w1f = frag16(brow, 32 + koff);
      float bias = sBias[nt * 16 + l16];
      #pragma unroll
      for (int mt = 0; mt < 2; ++mt) {
        v8f acc = {};
        acc = WMMA_F16(a[mt][0], w0,  acc);
        acc = WMMA_F16(a[mt][1], w1f, acc);
        _Float16* o = sB + (wave * 32 + mt * 16 + hi16 * 8) * SROW + nt * 16 + l16;
        #pragma unroll
        for (int rr = 0; rr < 8; ++rr) o[rr * SROW] = (_Float16)(acc[rr] + bias);
      }
    }
  }
  __syncthreads();

  // ---------------- RMSNorm + SiLU over sB rows (1 row / thread, vectorized) ----------------
  {
    _Float16* hrow = sB + tid * SROW;
    v8h ch[16];
    float ss = 0.0f;
    #pragma unroll
    for (int c = 0; c < 16; ++c) {
      ch[c] = *(const v8h*)(hrow + c * 8);
      #pragma unroll
      for (int e = 0; e < 8; ++e) { float x = (float)ch[c][e]; ss += x * x; }
    }
    float scale = rsqrtf(ss * (1.0f / 128.0f) + 1e-5f);
    #pragma unroll
    for (int c = 0; c < 16; ++c) {
      v8h o;
      #pragma unroll
      for (int e = 0; e < 8; ++e) {
        float x = (float)ch[c][e] * scale * sRms[c * 8 + e];
        o[e] = (_Float16)(x / (1.0f + __expf(-x)));
      }
      *(v8h*)(hrow + c * 8) = o;
    }
  }
  __syncthreads();

  // ---------------- Stage 2 weights: W2^T (coalesced), b2 ----------------
  for (int idx = tid; idx < 128 * 128; idx += THREADS) {
    int k = idx >> 7, n = idx & 127;
    sW[n * SROW + k] = (_Float16)W2[idx];     // W2[idx] == W2[k*128 + n]
  }
  for (int n = tid; n < 128; n += THREADS) sBias[n] = b2v[n];
  __syncthreads();

  // ---------------- GEMM2: h2 = h1(256x128) @ W2 + b2 -> sA (f16) ----------------
  {
    v16h a[2][4];
    #pragma unroll
    for (int mt = 0; mt < 2; ++mt) {
      const _Float16* arow = sB + (wave * 32 + mt * 16 + l16) * SROW;
      #pragma unroll
      for (int c = 0; c < 4; ++c) a[mt][c] = frag16(arow, c * 32 + koff);
    }
    #pragma unroll
    for (int nt = 0; nt < 8; ++nt) {
      const _Float16* brow = sW + (nt * 16 + l16) * SROW;
      v16h w[4];
      #pragma unroll
      for (int c = 0; c < 4; ++c) w[c] = frag16(brow, c * 32 + koff);
      float bias = sBias[nt * 16 + l16];
      #pragma unroll
      for (int mt = 0; mt < 2; ++mt) {
        v8f acc = {};
        #pragma unroll
        for (int c = 0; c < 4; ++c) acc = WMMA_F16(a[mt][c], w[c], acc);
        _Float16* o = sA + (wave * 32 + mt * 16 + hi16 * 8) * SROW + nt * 16 + l16;
        #pragma unroll
        for (int rr = 0; rr < 8; ++rr) o[rr * SROW] = (_Float16)(acc[rr] + bias);
      }
    }
  }
  __syncthreads();

  // ---------------- Stage 3 weights: Wa^T (coalesced), ba ----------------
  for (int idx = tid; idx < 128 * 256; idx += THREADS) {
    int k = idx >> 8, n = idx & 255;
    sW[n * SROW + k] = (_Float16)Wa[idx];     // Wa[idx] == Wa[k*256 + n]
  }
  for (int n = tid; n < 256; n += THREADS) sBias[n] = bav[n];
  __syncthreads();

  // ---------------- GEMM3: out = h2(256x128) @ Wa + ba -> global f32 ----------------
  {
    v16h a[2][4];
    #pragma unroll
    for (int mt = 0; mt < 2; ++mt) {
      const _Float16* arow = sA + (wave * 32 + mt * 16 + l16) * SROW;
      #pragma unroll
      for (int c = 0; c < 4; ++c) a[mt][c] = frag16(arow, c * 32 + koff);
    }
    const long long blockrow = (long long)blockIdx.x * ROWS + wave * 32;
    #pragma unroll
    for (int nt = 0; nt < 16; ++nt) {
      const _Float16* brow = sW + (nt * 16 + l16) * SROW;
      v16h w[4];
      #pragma unroll
      for (int c = 0; c < 4; ++c) w[c] = frag16(brow, c * 32 + koff);
      float bias = sBias[nt * 16 + l16];
      #pragma unroll
      for (int mt = 0; mt < 2; ++mt) {
        v8f acc = {};
        #pragma unroll
        for (int c = 0; c < 4; ++c) acc = WMMA_F16(a[mt][c], w[c], acc);
        float* o = out + (blockrow + mt * 16 + hi16 * 8) * 256 + nt * 16 + l16;
        #pragma unroll
        for (int rr = 0; rr < 8; ++rr) o[rr * 256] = acc[rr] + bias;
      }
    }
  }
}

extern "C" void kernel_launch(void* const* d_in, const int* in_sizes, int n_in,
                              void* d_out, int out_size, void* d_ws, size_t ws_size,
                              hipStream_t stream) {
  const float* pos   = (const float*)d_in[0];
  const float* vel   = (const float*)d_in[1];
  const float* att   = (const float*)d_in[2];
  const float* W1    = (const float*)d_in[3];
  const float* b1    = (const float*)d_in[4];
  const float* rms_w = (const float*)d_in[5];
  const float* W2    = (const float*)d_in[6];
  const float* b2    = (const float*)d_in[7];
  const float* Wa    = (const float*)d_in[8];
  const float* ba    = (const float*)d_in[9];
  float* out = (float*)d_out;

  // pos has B*T*N*2 elements; pairs = B*T*N*N (N = 32)
  long long agents = (long long)in_sizes[0] / 2;
  long long pairs  = agents * 32;
  int blocks = (int)(pairs / ROWS);

  relenc_fused<<<blocks, THREADS, 0, stream>>>(pos, vel, att, W1, b1, rms_w,
                                               W2, b2, Wa, ba, out);
}
